// NeuroSAT_87978110091274
// MI455X (gfx1250) — matrix operand order repaired and checked by reading, not compiled, not run on hardware
//
#include <hip/hip_runtime.h>
#include <hip/hip_bf16.h>

typedef __attribute__((ext_vector_type(16))) _Float16 v16h;
typedef __attribute__((ext_vector_type(8)))  float    v8f;

#define NVARS    20000
#define NLITS    40000
#define NCLAUSES 84000
#define NEDGES   400000
#define DIM      128
#define NROUNDS  8
#define NBATCH   40

// ---------------------------------------------------------------------------
// Pack a fp32 weight matrix B[K,N] (possibly given transposed: B[k][n]=src[n][k])
// into a WMMA f16 B-fragment layout (contiguous k, matching the A build) so the GEMM can read each
// fragment as one contiguous 32B load per lane.
//   fragment (kstep, ntile, lane): elem e -> B[kstep*32 + 16*(lane>>4) + e][ntile*16 + (lane&15)]
// ---------------------------------------------------------------------------
__global__ __launch_bounds__(256) void pack_b_kernel(
    const float* __restrict__ src, _Float16* __restrict__ dst,
    int Ncols, int k0, int kcnt, int trans)
{
    int idx = blockIdx.x * blockDim.x + threadIdx.x;
    int ntpc = Ncols >> 4;
    int total = (kcnt >> 5) * ntpc * 32;
    if (idx >= total) return;
    int lane  = idx & 31;
    int t     = idx >> 5;
    int ntile = t % ntpc;
    int ksl   = t / ntpc;              // kstep local to this source chunk
    int ksg   = (k0 >> 5) + ksl;       // kstep in the combined GEMM K space
    int n     = ntile * 16 + (lane & 15);
    int kb    = ksl * 32 + 16 * (lane >> 4);
    _Float16* o = dst + ((size_t)(ksg * ntpc + ntile) * 32 + lane) * 16;
#pragma unroll
    for (int e = 0; e < 16; e++) {
        int kk = kb + e;  // local k within source
        float v = trans ? src[(size_t)n * kcnt + kk] : src[(size_t)kk * Ncols + n];
        o[e] = (_Float16)v;
    }
}

// ---------------------------------------------------------------------------
// Y[Nrows,Ncols] = act(Xcat @ Bpacked + bias), where Xcat is the virtual
// column-concatenation of up to 3 pieces X0|X1|X2, each [Nrows,128] fp32
// (fuses the LSTM-input concat into the GEMM; piece 1 may be row-flipped:
// literal negation roll srow=(row+NVARS)%NLITS). K = nPieces*128.
// 8 waves/block; each wave: 16 rows x 64 cols (4 WMMA tiles), K-loop step 32.
// B fragments for all 4 tiles are issued before the A-fragment build so the
// weight loads overlap the activation-load latency.
// ---------------------------------------------------------------------------
__global__ __launch_bounds__(256) void gemm_wmma(
    const float* __restrict__ X0, const float* __restrict__ X1,
    const float* __restrict__ X2, int flip1,
    const _Float16* __restrict__ Bp, const float* __restrict__ bias,
    float* __restrict__ Y, int Nrows, int nPieces, int Ncols, int relu)
{
    const int lane  = threadIdx.x & 31;
    const int half  = lane >> 4;
    const int ln    = lane & 15;
    const int wid   = threadIdx.x >> 5;
    const int rbase = blockIdx.x * 128 + wid * 16;
    if (rbase >= Nrows) return;                 // wave-uniform: EXEC stays all-1s
    const int ntiles = Ncols >> 4;
    const int ct0    = blockIdx.y << 2;         // first of 4 column tiles
    const int row    = rbase + ln;
    const int frow   = flip1 ? (row < NVARS ? row + NVARS : row - NVARS) : row;
    const float* xr0 = X0 + (size_t)row * DIM;
    const float* xr1 = X1 ? X1 + (size_t)frow * DIM : nullptr;
    const float* xr2 = X2 ? X2 + (size_t)row * DIM : nullptr;
    v8f acc[4] = {};
    int ksg = 0;
    for (int p = 0; p < nPieces; p++) {
        const float* xr = (p == 0) ? xr0 : ((p == 1) ? xr1 : xr2);
#pragma unroll
        for (int kk = 0; kk < 4; kk++, ksg++) {
            // issue all 4 B-fragment loads first (single 32B load each)
            const v16h* bb = (const v16h*)Bp + (size_t)(ksg * ntiles + ct0) * 32 + lane;
            v16h b0 = bb[0 * 32];
            v16h b1 = bb[1 * 32];
            v16h b2 = bb[2 * 32];
            v16h b3 = bb[3 * 32];
            // A fragment: elem e -> K=16*half+e (matches packed B)
            const float* xp = xr + kk * 32 + 16 * half;
            v16h a;
#pragma unroll
            for (int e = 0; e < 8; e++) a[e]     = (_Float16)xp[e];
#pragma unroll
            for (int e = 0; e < 8; e++) a[8 + e] = (_Float16)xp[8 + e];
            acc[0] = __builtin_amdgcn_wmma_f32_16x16x32_f16(false, a, false, b0, (short)0, acc[0], false, false);
            acc[1] = __builtin_amdgcn_wmma_f32_16x16x32_f16(false, a, false, b1, (short)0, acc[1], false, false);
            acc[2] = __builtin_amdgcn_wmma_f32_16x16x32_f16(false, a, false, b2, (short)0, acc[2], false, false);
            acc[3] = __builtin_amdgcn_wmma_f32_16x16x32_f16(false, a, false, b3, (short)0, acc[3], false, false);
        }
    }
    // C/D layout: lane n = lane&15, VGPR v -> row m = v + 8*half
#pragma unroll
    for (int t = 0; t < 4; t++) {
        const int col = ((ct0 + t) << 4) + ln;
        const float bv = bias ? bias[col] : 0.0f;
#pragma unroll
        for (int v = 0; v < 8; v++) {
            float y = acc[t][v] + bv;
            if (relu) y = fmaxf(y, 0.0f);
            Y[(size_t)(rbase + v + 8 * half) * Ncols + col] = y;
        }
    }
}

// ---------------------------------------------------------------------------
__global__ __launch_bounds__(256) void init_state_kernel(
    float* __restrict__ Lh, float* __restrict__ Lc,
    float* __restrict__ Ch, float* __restrict__ Cc,
    const float* __restrict__ L0, const float* __restrict__ C0)
{
    size_t t = (size_t)blockIdx.x * blockDim.x + threadIdx.x;
    int d = (int)(t & (DIM - 1));
    size_t row = t >> 7;
    const float inv = 0.08838834764831845f;  // 1/sqrt(128)
    if (row < NLITS)    { Lh[t] = L0[d] * inv; Lc[t] = 0.0f; }
    if (row < NCLAUSES) { Ch[t] = C0[d] * inv; Cc[t] = 0.0f; }
}

__global__ __launch_bounds__(256) void zero_kernel(float* __restrict__ p, size_t n)
{
    size_t t = (size_t)blockIdx.x * blockDim.x + threadIdx.x;
    if (t < n) p[t] = 0.0f;
}

// segment_sum over edges: dst[didx[e]] += src[sidx[e]]   (32 lanes x float4 per edge)
__global__ __launch_bounds__(256) void scatter_add_kernel(
    const float* __restrict__ src, const int* __restrict__ sidx,
    const int* __restrict__ didx, float* __restrict__ dst, int E)
{
    int t = blockIdx.x * blockDim.x + threadIdx.x;
    int e = t >> 5;
    if (e >= E) return;
    int q = (t & 31) * 4;
    const float* s = src + (size_t)sidx[e] * DIM + q;
    float*       d = dst + (size_t)didx[e] * DIM + q;
    float4 v = *(const float4*)s;
    atomicAdd(d + 0, v.x); atomicAdd(d + 1, v.y);
    atomicAdd(d + 2, v.z); atomicAdd(d + 3, v.w);
}

// LayerNorm-LSTM cell update; one wave32 per row; gates[N,512] = [i|j|f|o]
__global__ __launch_bounds__(256) void lstm_kernel(
    const float* __restrict__ gates, float* __restrict__ h, float* __restrict__ c,
    const float* __restrict__ lng, const float* __restrict__ lnb, int N)
{
    int row  = (int)((blockIdx.x * (size_t)blockDim.x + threadIdx.x) >> 5);
    int lane = threadIdx.x & 31;
    if (row >= N) return;
    const float* g = gates + (size_t)row * 4 * DIM;
    const float eps = 1e-5f;
    float xg[4][4], mean[4], rstd[4];
#pragma unroll
    for (int ga = 0; ga < 4; ga++) {
        float s = 0.0f, s2 = 0.0f;
#pragma unroll
        for (int j = 0; j < 4; j++) {
            int d = lane + 32 * j;
            float v = g[ga * DIM + d];
            xg[ga][j] = v; s += v; s2 += v * v;
        }
#pragma unroll
        for (int o = 16; o >= 1; o >>= 1) { s += __shfl_xor(s, o, 32); s2 += __shfl_xor(s2, o, 32); }
        float m = s * (1.0f / DIM);
        mean[ga] = m;
        rstd[ga] = rsqrtf(s2 * (1.0f / DIM) - m * m + eps);
    }
    float c2v[4];
    float s = 0.0f, s2 = 0.0f;
#pragma unroll
    for (int j = 0; j < 4; j++) {
        int d = lane + 32 * j;
        float i_ = lng[0 * DIM + d] * (xg[0][j] - mean[0]) * rstd[0] + lnb[0 * DIM + d];
        float j_ = lng[1 * DIM + d] * (xg[1][j] - mean[1]) * rstd[1] + lnb[1 * DIM + d];
        float f_ = lng[2 * DIM + d] * (xg[2][j] - mean[2]) * rstd[2] + lnb[2 * DIM + d];
        float si = 1.0f / (1.0f + __expf(-i_));
        float sf = 1.0f / (1.0f + __expf(-f_));
        float c2 = sf * c[(size_t)row * DIM + d] + si * tanhf(j_);
        c2v[j] = c2; s += c2; s2 += c2 * c2;
    }
#pragma unroll
    for (int o = 16; o >= 1; o >>= 1) { s += __shfl_xor(s, o, 32); s2 += __shfl_xor(s2, o, 32); }
    float mc = s * (1.0f / DIM);
    float rc = rsqrtf(s2 * (1.0f / DIM) - mc * mc + eps);
#pragma unroll
    for (int j = 0; j < 4; j++) {
        int d = lane + 32 * j;
        float o_ = lng[3 * DIM + d] * (xg[3][j] - mean[3]) * rstd[3] + lnb[3 * DIM + d];
        float so = 1.0f / (1.0f + __expf(-o_));
        float cn = lng[4 * DIM + d] * (c2v[j] - mc) * rc + lnb[4 * DIM + d];
        c[(size_t)row * DIM + d] = c2v[j];
        h[(size_t)row * DIM + d] = so * tanhf(cn);
    }
}

// votes[row] = X[row,:] . Wout + bout   (one wave per row)
__global__ __launch_bounds__(256) void vote_out_kernel(
    const float* __restrict__ X, const float* __restrict__ Wout,
    const float* __restrict__ bout, float* __restrict__ votes, int N)
{
    int row  = (int)((blockIdx.x * (size_t)blockDim.x + threadIdx.x) >> 5);
    int lane = threadIdx.x & 31;
    if (row >= N) return;
    float s = 0.0f;
#pragma unroll
    for (int j = 0; j < 4; j++) {
        int d = lane + 32 * j;
        s += X[(size_t)row * DIM + d] * Wout[d];
    }
#pragma unroll
    for (int o = 16; o >= 1; o >>= 1) s += __shfl_xor(s, o, 32);
    if (lane == 0) votes[row] = s + bout[0];
}

// logits[b] = mean over 500 vars x 2 polarities + vote_bias
__global__ __launch_bounds__(256) void logits_kernel(
    const float* __restrict__ votes, const float* __restrict__ vote_bias,
    float* __restrict__ out)
{
    __shared__ float sh[8];
    const int per = NVARS / NBATCH;  // 500
    int b = blockIdx.x, tid = threadIdx.x;
    float s = 0.0f;
    for (int v = tid; v < per; v += 256) {
        int r = b * per + v;
        s += votes[r] + votes[r + NVARS];
    }
#pragma unroll
    for (int o = 16; o >= 1; o >>= 1) s += __shfl_xor(s, o, 32);
    if ((tid & 31) == 0) sh[tid >> 5] = s;
    __syncthreads();
    if (tid == 0) {
        float t = 0.0f;
        for (int i = 0; i < 8; i++) t += sh[i];
        out[b] = t / (float)(per * 2) + vote_bias[0];
    }
}

// ---------------------------------------------------------------------------
extern "C" void kernel_launch(void* const* d_in, const int* in_sizes, int n_in,
                              void* d_out, int out_size, void* d_ws, size_t ws_size,
                              hipStream_t stream)
{
    (void)in_sizes; (void)n_in; (void)out_size; (void)ws_size;

    const int*   lit_idx    = (const int*)d_in[0];
    const int*   clause_idx = (const int*)d_in[1];
    const float* L_init     = (const float*)d_in[2];
    const float* C_init     = (const float*)d_in[3];
    const float* LC_W       = (const float*)d_in[4];
    const float* LC_b       = (const float*)d_in[5];
    const float* CL_W       = (const float*)d_in[6];
    const float* CL_b       = (const float*)d_in[7];
    const float* Lu_Wih     = (const float*)d_in[8];
    const float* Lu_Whh     = (const float*)d_in[9];
    const float* Lu_lng     = (const float*)d_in[10];
    const float* Lu_lnb     = (const float*)d_in[11];
    const float* Cu_Wih     = (const float*)d_in[12];
    const float* Cu_Whh     = (const float*)d_in[13];
    const float* Cu_lng     = (const float*)d_in[14];
    const float* Cu_lnb     = (const float*)d_in[15];
    const float* vote_W     = (const float*)d_in[16];
    const float* vote_b     = (const float*)d_in[17];
    const float* vote_Wout  = (const float*)d_in[18];
    const float* vote_bout  = (const float*)d_in[19];
    const float* vote_bias  = (const float*)d_in[20];

    // ---- carve workspace (256B aligned regions) ----
    size_t off = 0;
    auto carve = [&](size_t bytes) {
        void* p = (char*)d_ws + off;
        off = (off + bytes + 255) & ~(size_t)255;
        return p;
    };
    _Float16* WLC   = (_Float16*)carve((size_t)4 * DIM * DIM * sizeof(_Float16));
    _Float16* WCL   = (_Float16*)carve((size_t)4 * DIM * DIM * sizeof(_Float16));
    _Float16* WCU   = (_Float16*)carve((size_t)256 * 512 * sizeof(_Float16));
    _Float16* WLU   = (_Float16*)carve((size_t)384 * 512 * sizeof(_Float16));
    _Float16* WVOTE = (_Float16*)carve((size_t)3 * DIM * DIM * sizeof(_Float16));
    float* Lh    = (float*)carve((size_t)NLITS    * DIM * sizeof(float));
    float* Lc    = (float*)carve((size_t)NLITS    * DIM * sizeof(float));
    float* Ch    = (float*)carve((size_t)NCLAUSES * DIM * sizeof(float));
    float* Cc    = (float*)carve((size_t)NCLAUSES * DIM * sizeof(float));
    float* bufA  = (float*)carve((size_t)NCLAUSES * DIM * sizeof(float));
    float* bufB  = (float*)carve((size_t)NCLAUSES * DIM * sizeof(float));
    float* msgs  = (float*)carve((size_t)NCLAUSES * DIM * sizeof(float));
    float* gates = (float*)carve((size_t)NCLAUSES * 512 * sizeof(float));
    float* votes = (float*)carve((size_t)NLITS * sizeof(float));

    auto packB = [&](const float* src, _Float16* dst, int Ncols, int k0, int kcnt, int trans) {
        int total = (kcnt / 32) * (Ncols / 16) * 32;
        pack_b_kernel<<<(total + 255) / 256, 256, 0, stream>>>(src, dst, Ncols, k0, kcnt, trans);
    };
    // plain GEMM: one input piece [N,128]
    auto gemm1 = [&](const float* X, const _Float16* Bp, const float* bias, float* Y,
                     int Nrows, int Ncols, int relu) {
        dim3 grid((Nrows + 127) / 128, Ncols / 64);
        gemm_wmma<<<grid, 256, 0, stream>>>(X, nullptr, nullptr, 0, Bp, bias, Y,
                                            Nrows, 1, Ncols, relu);
    };
    auto zero = [&](float* p, size_t n) {
        zero_kernel<<<(unsigned)((n + 255) / 256), 256, 0, stream>>>(p, n);
    };

    // ---- pack weights into WMMA B-fragment layout (fp32 -> f16) ----
    for (int i = 0; i < 4; i++) packB(LC_W + (size_t)i * DIM * DIM, WLC + (size_t)i * DIM * DIM, DIM, 0, DIM, 0);
    for (int i = 0; i < 4; i++) packB(CL_W + (size_t)i * DIM * DIM, WCL + (size_t)i * DIM * DIM, DIM, 0, DIM, 0);
    packB(Cu_Wih, WCU, 512, 0,   128, 1);   // rows 0..127   = Cu_Wih.T
    packB(Cu_Whh, WCU, 512, 128, 128, 1);   // rows 128..255 = Cu_Whh.T
    packB(Lu_Wih, WLU, 512, 0,   256, 1);   // rows 0..255   = Lu_Wih.T
    packB(Lu_Whh, WLU, 512, 256, 128, 1);   // rows 256..383 = Lu_Whh.T
    for (int i = 0; i < 3; i++) packB(vote_W + (size_t)i * DIM * DIM, WVOTE + (size_t)i * DIM * DIM, DIM, 0, DIM, 0);

    // ---- init states ----
    init_state_kernel<<<(unsigned)(((size_t)NCLAUSES * DIM + 255) / 256), 256, 0, stream>>>(
        Lh, Lc, Ch, Cc, L_init, C_init);

    // ---- message-passing rounds ----
    for (int r = 0; r < NROUNDS; r++) {
        // literal -> clause: 4-layer MLP on Lh
        gemm1(Lh,   WLC + 0 * DIM * DIM, LC_b + 0 * DIM, bufA, NLITS, DIM, 1);
        gemm1(bufA, WLC + 1 * DIM * DIM, LC_b + 1 * DIM, bufB, NLITS, DIM, 1);
        gemm1(bufB, WLC + 2 * DIM * DIM, LC_b + 2 * DIM, bufA, NLITS, DIM, 1);
        gemm1(bufA, WLC + 3 * DIM * DIM, LC_b + 3 * DIM, bufB, NLITS, DIM, 0);
        zero(msgs, (size_t)NCLAUSES * DIM);
        scatter_add_kernel<<<(NEDGES * 32 + 255) / 256, 256, 0, stream>>>(
            bufB, lit_idx, clause_idx, msgs, NEDGES);
        // clause LN-LSTM: gates = [msgs|Ch] @ [Wih.T;Whh.T]   (concat fused into GEMM)
        {
            dim3 grid((NCLAUSES + 127) / 128, 512 / 64);
            gemm_wmma<<<grid, 256, 0, stream>>>(msgs, Ch, nullptr, 0, WCU, nullptr,
                                                gates, NCLAUSES, 2, 512, 0);
        }
        lstm_kernel<<<(NCLAUSES + 7) / 8, 256, 0, stream>>>(gates, Ch, Cc, Cu_lng, Cu_lnb, NCLAUSES);

        // clause -> literal: 4-layer MLP on Ch
        gemm1(Ch,   WCL + 0 * DIM * DIM, CL_b + 0 * DIM, bufA, NCLAUSES, DIM, 1);
        gemm1(bufA, WCL + 1 * DIM * DIM, CL_b + 1 * DIM, bufB, NCLAUSES, DIM, 1);
        gemm1(bufB, WCL + 2 * DIM * DIM, CL_b + 2 * DIM, bufA, NCLAUSES, DIM, 1);
        gemm1(bufA, WCL + 3 * DIM * DIM, CL_b + 3 * DIM, bufB, NCLAUSES, DIM, 0);
        zero(msgs, (size_t)NLITS * DIM);
        scatter_add_kernel<<<(NEDGES * 32 + 255) / 256, 256, 0, stream>>>(
            bufB, clause_idx, lit_idx, msgs, NEDGES);
        // literal LN-LSTM: gates = [msgs|flip(Lh)|Lh] @ [Wih.T;Whh.T] (concat+flip fused)
        {
            dim3 grid((NLITS + 127) / 128, 512 / 64);
            gemm_wmma<<<grid, 256, 0, stream>>>(msgs, Lh, Lh, 1, WLU, nullptr,
                                                gates, NLITS, 3, 512, 0);
        }
        lstm_kernel<<<(NLITS + 7) / 8, 256, 0, stream>>>(gates, Lh, Lc, Lu_lng, Lu_lnb, NLITS);
    }

    // ---- vote head ----
    gemm1(Lh,   WVOTE + 0 * DIM * DIM, vote_b + 0 * DIM, bufA, NLITS, DIM, 1);
    gemm1(bufA, WVOTE + 1 * DIM * DIM, vote_b + 1 * DIM, bufB, NLITS, DIM, 1);
    gemm1(bufB, WVOTE + 2 * DIM * DIM, vote_b + 2 * DIM, bufA, NLITS, DIM, 1);
    vote_out_kernel<<<(NLITS * 32 + 255) / 256, 256, 0, stream>>>(
        bufA, vote_Wout, vote_bout, votes, NLITS);
    logits_kernel<<<NBATCH, 256, 0, stream>>>(votes, vote_bias, (float*)d_out);
}
